// FFNNTransducerModel_31001073942673
// MI455X (gfx1250) — matrix operand order repaired and checked by reading, not compiled, not run on hardware
//
#include <hip/hip_runtime.h>
#include <math.h>

typedef __attribute__((ext_vector_type(16))) _Float16 v16h;
typedef __attribute__((ext_vector_type(8)))  _Float16 v8h;
typedef __attribute__((ext_vector_type(8)))  float    v8f;

#define B_   8
#define T_   1000
#define TP_  1008   // padded T rows in workspace (63 * 16)
#define E_   512
#define U_   100
#define UP1  101
#define D_   256
#define P_   256
#define J_   128
#define V_   88
#define UT_  7      // ceil(101/16) u-tiles
#define NT_  6      // ceil(88/16) v-tiles

// workspace byte offsets
#define PP_BYTES    (B_ * UP1 * J_ * 4)            // 413,696
#define ENCB_BYTES  (B_ * TP_ * J_ * 4)            // 4,128,768
#define JW1F_HALVES (16 * 8 * 32 * 16)             // 65,536
#define JW2F_HALVES (4 * 6 * 32 * 16)              // 12,288

// fast tanh: 1 - 2/(exp(2x)+1)  -> single v_exp_f32 transcendental
__device__ __forceinline__ float fast_tanh(float x) {
    const float e = __expf(2.0f * x);
    return 1.0f - 2.0f / (e + 1.0f);
}

// ---------------------------------------------------------------------------
// Kernel 0: pre-swizzle jw1[:E] and jw2 into per-lane WMMA B-fragment order.
// Fragment layout: [frag][lane(32)][i(16)] halves -> 32B contiguous per lane.
// ---------------------------------------------------------------------------
__global__ void prep_frag_kernel(const float* __restrict__ jw1,
                                 const float* __restrict__ jw2,
                                 _Float16* __restrict__ jw1f,
                                 _Float16* __restrict__ jw2f) {
    const int tid = threadIdx.x;                    // 256 threads
    for (int idx = tid; idx < JW1F_HALVES; idx += 256) {
        const int i    = idx & 15;
        const int lane = (idx >> 4) & 31;
        const int f    = idx >> 9;                  // kk*8 + nt
        const int kk   = f >> 3, nt = f & 7;
        const int k    = kk * 32 + ((lane >> 4) << 3) + (i & 7) + ((i >> 3) << 4);   // dense-B K striping
        const int n    = nt * 16 + (lane & 15);
        jw1f[idx] = (_Float16)jw1[k * J_ + n];
    }
    for (int idx = tid; idx < JW2F_HALVES; idx += 256) {
        const int i    = idx & 15;
        const int lane = (idx >> 4) & 31;
        const int f    = idx >> 9;                  // kk*6 + nt
        const int kk   = f / 6, nt = f - kk * 6;
        const int k    = kk * 32 + ((lane >> 4) << 3) + (i & 7) + ((i >> 3) << 4);
        const int n    = nt * 16 + (lane & 15);
        const float w  = jw2[k * V_ + ((n < V_) ? n : (V_ - 1))];
        jw2f[idx] = (_Float16)((n < V_) ? w : 0.f);
    }
}

// ---------------------------------------------------------------------------
// Kernel 1: prediction network -> pred_proj[B,101,128] (fp32, workspace)
// ---------------------------------------------------------------------------
__global__ void pred_net_kernel(const int* __restrict__ targets,
                                const float* __restrict__ emb,
                                const float* __restrict__ pw1,
                                const float* __restrict__ pb1,
                                const float* __restrict__ pw2,
                                const float* __restrict__ pb2,
                                const float* __restrict__ jw1,
                                float* __restrict__ pp) {
    __shared__ float s_e[2 * D_];
    __shared__ float s_h[P_];
    __shared__ float s_p[P_];
    const int b = blockIdx.x;
    const int p = threadIdx.x;          // 0..255
    for (int u = 0; u <= U_; ++u) {
        const int tok0 = (u >= 1) ? targets[b * U_ + (u - 1)] : (V_ - 1);
        const int tok1 = (u >= 2) ? targets[b * U_ + (u - 2)] : (V_ - 1);
        for (int idx = p; idx < 2 * D_; idx += 256) {
            const int tok = (idx < D_) ? tok0 : tok1;
            s_e[idx] = emb[tok * D_ + (idx & (D_ - 1))];
        }
        __syncthreads();
        float a1 = pb1[p];
        for (int k = 0; k < 2 * D_; ++k) a1 += s_e[k] * pw1[k * P_ + p];
        s_h[p] = tanhf(a1);
        __syncthreads();
        float a2 = pb2[p];
        for (int k = 0; k < P_; ++k) a2 += s_h[k] * pw2[k * P_ + p];
        s_p[p] = tanhf(a2);
        __syncthreads();
        if (p < J_) {
            float a3 = 0.f;
            for (int k = 0; k < P_; ++k) a3 += s_p[k] * jw1[(E_ + k) * J_ + p];
            pp[((long)b * UP1 + u) * J_ + p] = a3;
        }
        __syncthreads();
    }
}

// ---------------------------------------------------------------------------
// Kernel 2: enc_proj_biased[B,1008,128] = encoder @ jw1[:E] + jb1  (WMMA f16)
// grid (63, B), block 128 (4 waves); wave w owns n-tiles {2w, 2w+1}.
// Input rows clamped, output rows padded -> zero predicated memory ops.
// ---------------------------------------------------------------------------
__global__ void enc_proj_kernel(const float* __restrict__ enc,
                                const v16h* __restrict__ jw1f,
                                const float* __restrict__ jb1,
                                float* __restrict__ enc_b) {
    const int b       = blockIdx.y;
    const int t0      = blockIdx.x * 16;
    const int lane    = threadIdx.x & 31;
    const int wave    = threadIdx.x >> 5;
    const int lane_lo = lane & 15;
    const int lane_hi = lane >> 4;

    v8f acc[2];
    #pragma unroll
    for (int s = 0; s < 2; ++s) {
        const int n = (wave * 2 + s) * 16 + lane_lo;
        const float bias = jb1[n];
        #pragma unroll
        for (int j = 0; j < 8; ++j) acc[s][j] = bias;   // fold jb1 into C init
    }

    int t_a = t0 + lane_lo;
    if (t_a >= T_) t_a = T_ - 1;                        // clamp: unconditional loads
    const float* erow = enc + ((long)b * T_ + t_a) * E_;

    for (int kk = 0; kk < E_ / 32; ++kk) {
        const int k0 = kk * 32 + lane_hi * 8;           // two 8-elem runs per lane
        float ev[16];
        *(float4*)&ev[0]  = *(const float4*)(erow + k0);
        *(float4*)&ev[4]  = *(const float4*)(erow + k0 + 4);
        *(float4*)&ev[8]  = *(const float4*)(erow + k0 + 16);
        *(float4*)&ev[12] = *(const float4*)(erow + k0 + 20);
        v16h af;
        #pragma unroll
        for (int i = 0; i < 16; ++i) af[i] = (_Float16)ev[i];

        #pragma unroll
        for (int s = 0; s < 2; ++s) {
            const v16h bf = jw1f[(kk * 8 + wave * 2 + s) * 32 + lane]; // 32B contiguous
            acc[s] = __builtin_amdgcn_wmma_f32_16x16x32_f16(
                false, af, false, bf, (short)0, acc[s], false, false);
        }
    }
    // unconditional store into padded buffer: base + immediate offsets
    float* orow = enc_b + ((long)b * TP_ + t0 + lane_hi * 8) * J_;
    #pragma unroll
    for (int s = 0; s < 2; ++s) {
        const int n = (wave * 2 + s) * 16 + lane_lo;
        #pragma unroll
        for (int j = 0; j < 8; ++j) orow[j * J_ + n] = acc[s][j];
    }
}

// ---------------------------------------------------------------------------
// Kernel 3: fused joint. grid (63, B), block 512 = 16 waves, one t per wave.
// out[b,t,u,v] = mask * (tanh(enc_b[t]+pp[u]) @ jw2 + jb2)
// ---------------------------------------------------------------------------
__global__ void __launch_bounds__(512)
joint_kernel(const float* __restrict__ enc_b, const float* __restrict__ pp,
             const float* __restrict__ jw2f_g, const float* __restrict__ jb2,
             const int* __restrict__ es,       const int* __restrict__ ts,
             float* __restrict__ out) {
    __shared__ v16h     s_bfrag[24 * 32];     // 24 KB: (kk*6+nt, lane) B fragments
    __shared__ _Float16 s_pp[112 * J_];       // 28 KB: pred_proj (f16, zero-padded u)
    __shared__ float    s_enc[16 * J_];       // 8 KB : per-wave encoder rows
    __shared__ float    s_jb2[NT_ * 16];      // 384 B: zero-padded joiner bias

    const int b   = blockIdx.y;
    const int tid = threadIdx.x;

    // coalesced vector copy of pre-swizzled jw2 fragments into LDS
    {
        const float4* src = (const float4*)jw2f_g;
        float4* dst = (float4*)s_bfrag;
        #pragma unroll
        for (int idx = tid; idx < (24 * 32 * 32) / 16; idx += 512) dst[idx] = src[idx];
    }
    // stage pred_proj as f16 (clamped load + value select: no predicated loads)
    for (int idx = tid; idx < 112 * J_; idx += 512) {
        const int u  = idx >> 7;
        const int uc = (u < UP1) ? u : (UP1 - 1);
        const float x = pp[((long)b * UP1 + uc) * J_ + (idx & (J_ - 1))];
        s_pp[idx] = (_Float16)((u < UP1) ? x : 0.f);
    }
    if (tid < NT_ * 16) {
        const float x = jb2[(tid < V_) ? tid : (V_ - 1)];
        s_jb2[tid] = (tid < V_) ? x : 0.f;
    }
    __syncthreads();

    const int wave = tid >> 5, lane = tid & 31;
    const int lane_lo = lane & 15, lane_hi = lane >> 4;
    const int t = blockIdx.x * 16 + wave;
    if (t >= T_) return;                               // wave-uniform

    {   // stage this wave's biased encoder projection row (128 f32)
        const float4 v = *(const float4*)&enc_b[(((long)b * TP_ + t) * J_) + lane * 4];
        *(float4*)&s_enc[wave * J_ + lane * 4] = v;
        const int tn = (t + 16 < T_) ? (t + 16) : t;
        __builtin_prefetch(&enc_b[(((long)b * TP_ + tn) * J_)], 0, 1);
    }
    const int  es_b    = es[b];
    const int  us1     = ts[b] + 1;
    const bool t_valid = (t < es_b);
    float* outbt = out + (((long)b * T_ + t) * UP1) * V_;

    float bias[NT_];
    #pragma unroll
    for (int nt = 0; nt < NT_; ++nt) bias[nt] = s_jb2[nt * 16 + lane_lo];

    const float* er = &s_enc[wave * J_];

    for (int ut = 0; ut < UT_; ++ut) {
        v8f acc[NT_];
        #pragma unroll
        for (int nt = 0; nt < NT_; ++nt)
            #pragma unroll
            for (int j = 0; j < 8; ++j) acc[nt][j] = bias[nt];

        const _Float16* ppr = &s_pp[(ut * 16 + lane_lo) * J_];
        #pragma unroll
        for (int kk = 0; kk < 4; ++kk) {
            const int k0 = kk * 32 + lane_hi * 8;      // two contiguous 8-elem runs
            float ev[16];
            *(float4*)&ev[0]  = *(const float4*)(er + k0);
            *(float4*)&ev[4]  = *(const float4*)(er + k0 + 4);
            *(float4*)&ev[8]  = *(const float4*)(er + k0 + 16);
            *(float4*)&ev[12] = *(const float4*)(er + k0 + 20);
            const v8h p0 = *(const v8h*)(ppr + k0);
            const v8h p1 = *(const v8h*)(ppr + k0 + 16);
            v16h af;
            #pragma unroll
            for (int i = 0; i < 16; ++i) {
                const float x = ev[i] + (float)((i < 8) ? p0[i] : p1[i - 8]);
                af[i] = (_Float16)fast_tanh(x);
            }
            #pragma unroll
            for (int nt = 0; nt < NT_; ++nt) {
                acc[nt] = __builtin_amdgcn_wmma_f32_16x16x32_f16(
                    false, af, false, s_bfrag[(kk * 6 + nt) * 32 + lane],
                    (short)0, acc[nt], false, false);
            }
        }
        // ---- store: full tiles have no u-guard (wave-uniform branch) ----
        if (ut < UT_ - 1) {
            const int u0 = ut * 16 + lane_hi * 8;
            float* orow = outbt + u0 * V_;
            #pragma unroll
            for (int nt = 0; nt < NT_; ++nt) {
                const int v = nt * 16 + lane_lo;
                if (v < V_) {                          // folds away for nt<=4
                    #pragma unroll
                    for (int j = 0; j < 8; ++j) {
                        const float val = (t_valid && (u0 + j < us1)) ? acc[nt][j] : 0.f;
                        orow[j * V_ + v] = val;
                    }
                }
            }
        } else {                                       // last tile: u in [96,112)
            const int u0 = (UT_ - 1) * 16 + lane_hi * 8;
            #pragma unroll
            for (int nt = 0; nt < NT_; ++nt) {
                const int v = nt * 16 + lane_lo;
                if (v >= V_) continue;
                #pragma unroll
                for (int j = 0; j < 8; ++j) {
                    const int u = u0 + j;
                    if (u < UP1) {
                        const float val = (t_valid && (u < us1)) ? acc[nt][j] : 0.f;
                        outbt[u * V_ + v] = val;
                    }
                }
            }
        }
    }
}

// ---------------------------------------------------------------------------
extern "C" void kernel_launch(void* const* d_in, const int* in_sizes, int n_in,
                              void* d_out, int out_size, void* d_ws, size_t ws_size,
                              hipStream_t stream) {
    const float* enc  = (const float*)d_in[0];
    const int*   es   = (const int*)  d_in[1];
    const int*   tg   = (const int*)  d_in[2];
    const int*   tsz  = (const int*)  d_in[3];
    const float* emb  = (const float*)d_in[4];
    const float* pw1  = (const float*)d_in[5];
    const float* pb1  = (const float*)d_in[6];
    const float* pw2  = (const float*)d_in[7];
    const float* pb2  = (const float*)d_in[8];
    const float* jw1  = (const float*)d_in[9];
    const float* jb1  = (const float*)d_in[10];
    const float* jw2  = (const float*)d_in[11];
    const float* jb2  = (const float*)d_in[12];
    float* out = (float*)d_out;

    // workspace layout (~4.7 MB): pp | enc_b(padded) | jw1 frags | jw2 frags
    char* w = (char*)d_ws;
    float*    pp    = (float*)w;
    float*    enc_b = (float*)(w + PP_BYTES);
    _Float16* jw1f  = (_Float16*)(w + PP_BYTES + ENCB_BYTES);
    _Float16* jw2f  = (_Float16*)(w + PP_BYTES + ENCB_BYTES + JW1F_HALVES * 2);

    prep_frag_kernel<<<dim3(1), dim3(256), 0, stream>>>(jw1, jw2, jw1f, jw2f);
    pred_net_kernel<<<dim3(B_), dim3(256), 0, stream>>>(tg, emb, pw1, pb1, pw2, pb2, jw1, pp);
    enc_proj_kernel<<<dim3(TP_ / 16, B_), dim3(128), 0, stream>>>(enc, (const v16h*)jw1f, jb1, enc_b);
    joint_kernel<<<dim3(TP_ / 16, B_), dim3(512), 0, stream>>>(enc_b, pp, (const float*)jw2f,
                                                               jb2, es, tsz, out);
}
